// SplitAttention_79001628443367
// MI455X (gfx1250) — matrix operand order repaired and checked by reading, not compiled, not run on hardware
//
#include <hip/hip_runtime.h>

// ---------------------------------------------------------------------------
// CDNA5 (gfx1250) fused split-attention:
//   proj:  qkv = x @ w_qkv ; qk_pos = pos @ w_qk_pos   (bf16 out, [b,h,n,64],
//          Q/Qpos pre-scaled by 0.5/sqrt(768))
//   attn:  flash attention with (content+positional) scores, online softmax.
//          K/Kpos tiles staged by the Tensor Data Mover (tensor_load_to_lds,
//          TENSORcnt), with TDM LDS padding matching the padded LDS layout.
//   outp:  out = attn_out @ w_out + b_out  (f32), A tile via ASYNC DMA.
// All matrix math via v_wmma_f32_16x16x32_bf16 (wave32).
// ---------------------------------------------------------------------------

typedef unsigned short u16;
typedef __attribute__((ext_vector_type(16))) __bf16 v16bf;
typedef __attribute__((ext_vector_type(8)))  float  v8f;
typedef __attribute__((ext_vector_type(4)))  unsigned int u32x4;
typedef __attribute__((ext_vector_type(8)))  int    i32x8;
typedef __attribute__((ext_vector_type(4)))  int    i32x4;

union Frag {                 // one WMMA 16x16x32 bf16 operand (8 VGPRs / lane)
    v16bf v;
    u32x4 q[2];
};

union H8 {                   // 16 bytes viewed as 8 bf16 raw halves
    u32x4 q;
    u16   e[8];
};

#define HEADS     12
#define DIM_HEAD  64
#define SEQ       1024
#define BATCH     4
#define DIM       768
#define INNER     768

#define Q_SCALE   0.01804219590264154f   /* 0.5 / sqrt(768) */

__device__ __forceinline__ u16 f2bf(float f) {
    __bf16 h = (__bf16)f;                       // hardware RNE convert
    return __builtin_bit_cast(unsigned short, h);
}

__device__ __forceinline__ v8f wmma_bf16(const Frag& a, const Frag& b, v8f c) {
    return __builtin_amdgcn_wmma_f32_16x16x32_bf16(
        false, a.v, false, b.v, (short)0, c, false, false);
}

// async global -> LDS copy, 16 bytes per lane, tracked by ASYNCcnt
__device__ __forceinline__ void async_copy_b128(unsigned lds_off, const void* gptr) {
    asm volatile("global_load_async_to_lds_b128 %0, %1, off"
                 :: "v"(lds_off), "v"((unsigned long long)(size_t)gptr)
                 : "memory");
}
__device__ __forceinline__ void wait_async0() {
    asm volatile("s_wait_asynccnt 0" ::: "memory");
}

// ---------------------------------------------------------------------------
// Tensor Data Mover: load a 64x64-element (bf16) 2-D tile, row stride 64
// elements in memory, into LDS with 16B padding after every 128B row
// (matching u16[64][72] layout). D# packed per CDNA5 ISA 8.3/8.4.
// ---------------------------------------------------------------------------
__device__ __forceinline__ void tdm_load_tile_64x64(unsigned lds_base, const void* gptr) {
    const unsigned long long ga = (unsigned long long)(size_t)gptr;
    u32x4 g0;
    g0[0] = 1u;                                   // count=1, user descriptor
    g0[1] = lds_base;                             // lds_addr (bytes)
    g0[2] = (unsigned)ga;                         // global_addr[31:0]
    g0[3] = (unsigned)((ga >> 32) & 0x1FFFFFFu)   // global_addr[56:32]
          | (2u << 30);                           // type = 2 ("image")
    i32x8 g1;
    g1[0] = (int)((1u << 16)      // data_size = 1 -> 2 bytes
                | (1u << 20)      // pad_enable
                | (4u << 22)      // pad_interval: 32 DWORDs (128B) between pads
                | (3u << 25));    // pad_amount: 4 DWORDs (16B)
    g1[1] = (int)(64u << 16);     // tensor_dim0 = 64 (elements)
    g1[2] = (int)(64u << 16);     // tensor_dim1 = 64
    g1[3] = (int)(64u << 16);     // tile_dim0 = 64
    g1[4] = 64;                   // tile_dim1 = 64, tile_dim2 = 0
    g1[5] = 64;                   // tensor_dim0_stride = 64 elements
    g1[6] = 0;                    // dim0_stride hi / dim1_stride lo
    g1[7] = 0;                    // dim1_stride hi (unused for 2-D)
    const i32x4 gz = {0, 0, 0, 0};
#if defined(__clang_major__) && (__clang_major__ >= 23)
    const i32x8 gz8 = {0, 0, 0, 0, 0, 0, 0, 0};
    __builtin_amdgcn_tensor_load_to_lds(g0, g1, gz, gz, gz8, 0);
#else
    __builtin_amdgcn_tensor_load_to_lds(g0, g1, gz, gz, 0);
#endif
}

// ---------------------------------------------------------------------------
// Kernel 1: fused QKV + positional projection.
// grid.x = 32 row tiles (128 rows), grid.y = 30 col tiles (18 qkv + 12 pos).
// Block tile 128x128, 8 waves, each wave a 32x64 sub-tile (2x4 WMMA accums).
// ---------------------------------------------------------------------------
__global__ void __launch_bounds__(256)
proj_kernel(const float* __restrict__ x, const float* __restrict__ pos,
            const float* __restrict__ w_qkv, const float* __restrict__ w_pos,
            u16* __restrict__ qw, u16* __restrict__ kw, u16* __restrict__ vw,
            u16* __restrict__ qpw, u16* __restrict__ kpw)
{
    __shared__ u16 As[128][40];   // [row][k]  stride 80B (16B aligned)
    __shared__ u16 Bs[128][40];   // [col][k]  (weights pre-transposed)

    const int ct  = blockIdx.y;
    const bool is_qkv = (ct < 18);
    const float* A = is_qkv ? x : pos;
    const float* W = is_qkv ? w_qkv : w_pos;
    const int ncols = is_qkv ? 2304 : 1536;
    const int c0    = is_qkv ? ct * 128 : (ct - 18) * 128;
    const int r0    = blockIdx.x * 128;

    const int tid  = threadIdx.x;
    const int lane = tid & 31;
    const int wave = tid >> 5;
    const int wrow = wave >> 1;       // 0..3 -> 32-row strip
    const int wcol = wave & 1;        // 0..1 -> 64-col strip
    const int m    = lane & 15;
    const int hi   = lane >> 4;

    v8f acc[2][4];
    const v8f zero = {0.f,0.f,0.f,0.f,0.f,0.f,0.f,0.f};
#pragma unroll
    for (int i = 0; i < 2; ++i)
#pragma unroll
        for (int j = 0; j < 4; ++j) acc[i][j] = zero;

    for (int k0 = 0; k0 < DIM; k0 += 32) {
        // stage A tile 128x32 (f32 -> bf16)
        {
            const int row = tid >> 1;
            const int kk  = (tid & 1) * 16;
            const float* src = &A[(size_t)(r0 + row) * DIM + k0 + kk];
#pragma unroll
            for (int j = 0; j < 16; ++j) As[row][kk + j] = f2bf(src[j]);
        }
        // stage B tile 32x128 transposed -> Bs[col][k]
        {
            const int kk = tid >> 3;
            const int n  = (tid & 7) * 16;
            const float* src = &W[(size_t)(k0 + kk) * ncols + c0 + n];
#pragma unroll
            for (int j = 0; j < 16; ++j) Bs[n + j][kk] = f2bf(src[j]);
        }
        __syncthreads();

        Frag af[2];
#pragma unroll
        for (int i = 0; i < 2; ++i) {
            const int ar = wrow * 32 + i * 16 + m;
            af[i].q[0] = *(const u32x4*)&As[ar][8 * hi];
            af[i].q[1] = *(const u32x4*)&As[ar][16 + 8 * hi];
        }
#pragma unroll
        for (int j = 0; j < 4; ++j) {
            Frag bf;
            const int bn = wcol * 64 + j * 16 + m;
            bf.q[0] = *(const u32x4*)&Bs[bn][8 * hi];
            bf.q[1] = *(const u32x4*)&Bs[bn][16 + 8 * hi];
#pragma unroll
            for (int i = 0; i < 2; ++i) acc[i][j] = wmma_bf16(af[i], bf, acc[i][j]);
        }
        __syncthreads();
    }

    // scatter-store into [b,h,n,64] bf16 tensors (Q/Qpos pre-scaled)
#pragma unroll
    for (int j = 0; j < 4; ++j) {
        const int c  = c0 + wcol * 64 + j * 16 + m;
        const int t  = c / INNER;
        const int cc = c - t * INNER;
        const int h  = cc >> 6;
        const int d  = cc & 63;
        u16* dst = is_qkv ? (t == 0 ? qw : (t == 1 ? kw : vw))
                          : (t == 0 ? qpw : kpw);
        const float scl = (t == 0) ? Q_SCALE : 1.0f;
#pragma unroll
        for (int i = 0; i < 2; ++i) {
#pragma unroll
            for (int vv = 0; vv < 8; ++vv) {
                const int r = r0 + wrow * 32 + i * 16 + vv + 8 * hi;
                const int b = r >> 10, n = r & 1023;
                dst[((((size_t)b * HEADS + h) << 10) + n) * DIM_HEAD + d] =
                    f2bf(acc[i][j][vv] * scl);
            }
        }
    }
}

// ---------------------------------------------------------------------------
// Kernel 2: flash attention. grid = (8 query tiles, B*H). 8 waves per block,
// each wave owns a 16-row query strip; 64-key tiles staged in LDS
// (K/Kpos via Tensor Data Mover, V transposed through registers).
// ---------------------------------------------------------------------------
__global__ void __launch_bounds__(256)
attn_kernel(const u16* __restrict__ qw, const u16* __restrict__ kw,
            const u16* __restrict__ vw, const u16* __restrict__ qpw,
            const u16* __restrict__ kpw, u16* __restrict__ aow)
{
    __shared__ u16 Ks [64][72];      // [key][d]   stride 144B (TDM-padded)
    __shared__ u16 Kps[64][72];
    __shared__ u16 Vt [64][72];      // [d][key]   (V transposed)
    __shared__ u16 Ps [8][16][40];   // per-wave P staging, stride 80B

    const int qt = blockIdx.x;                 // 0..7
    const int bh = blockIdx.y;                 // 0..47
    const int b  = bh / HEADS, h = bh % HEADS;

    const int tid  = threadIdx.x;
    const int lane = tid & 31;
    const int wave = tid >> 5;
    const int m    = lane & 15;
    const int hi   = lane >> 4;

    const size_t head_base = ((size_t)(b * HEADS + h)) << 10;   // * SEQ

    // load Q / Qpos fragments (two 32-wide d-chunks each); pre-scaled by proj
    Frag aq[2], aqp[2];
    {
        const int qrow = qt * 128 + wave * 16 + m;
        const size_t qb = (head_base + qrow) * DIM_HEAD;
#pragma unroll
        for (int dc = 0; dc < 2; ++dc) {
            aq[dc].q[0]  = *(const u32x4*)&qw [qb + dc * 32 + 8 * hi];
            aq[dc].q[1]  = *(const u32x4*)&qw [qb + dc * 32 + 16 + 8 * hi];
            aqp[dc].q[0] = *(const u32x4*)&qpw[qb + dc * 32 + 8 * hi];
            aqp[dc].q[1] = *(const u32x4*)&qpw[qb + dc * 32 + 16 + 8 * hi];
        }
    }

    const v8f zero = {0.f,0.f,0.f,0.f,0.f,0.f,0.f,0.f};
    v8f o[4];
#pragma unroll
    for (int dj = 0; dj < 4; ++dj) o[dj] = zero;
    float m_run[8], l_run[8];
#pragma unroll
    for (int vv = 0; vv < 8; ++vv) { m_run[vv] = -1e30f; l_run[vv] = 0.f; }

    const int krow = tid >> 2;          // 0..63
    const int dcol = (tid & 3) * 16;    // 0,16,32,48

    for (int kt = 0; kt < SEQ / 64; ++kt) {
        const size_t tile_base = (head_base + (size_t)kt * 64) * DIM_HEAD;
        // K / Kpos tiles: one TDM descriptor each, issued by wave 0 only
        if (wave == 0) {
            tdm_load_tile_64x64((unsigned)(size_t)&Ks[0][0],  &kw[tile_base]);
            tdm_load_tile_64x64((unsigned)(size_t)&Kps[0][0], &kpw[tile_base]);
        }
        // V^T tile via registers (TDM cannot transpose)
        {
            const size_t sb = tile_base + (size_t)krow * DIM_HEAD + dcol;
            H8 va, vb;
            va.q = *(const u32x4*)&vw[sb];
            vb.q = *(const u32x4*)&vw[sb + 8];
#pragma unroll
            for (int j = 0; j < 8; ++j) {
                Vt[dcol + j][krow]     = va.e[j];
                Vt[dcol + 8 + j][krow] = vb.e[j];
            }
            if (kt + 1 < SEQ / 64) {    // prefetch next tile
                __builtin_prefetch(&kw [sb + 64 * DIM_HEAD], 0, 1);
                __builtin_prefetch(&kpw[sb + 64 * DIM_HEAD], 0, 1);
                __builtin_prefetch(&vw [sb + 64 * DIM_HEAD], 0, 1);
            }
        }
        __builtin_amdgcn_s_wait_tensorcnt(0);   // no-op for non-issuing waves
        __syncthreads();

        for (int kc = 0; kc < 2; ++kc) {       // 32-key chunks
            const int kb = kc * 32;
            v8f s[2]; s[0] = zero; s[1] = zero;
#pragma unroll
            for (int sub = 0; sub < 2; ++sub) {
                const int key = kb + sub * 16 + m;
                Frag kf;
#pragma unroll
                for (int dc = 0; dc < 2; ++dc) {
                    kf.q[0] = *(const u32x4*)&Ks[key][dc * 32 + 8 * hi];
                    kf.q[1] = *(const u32x4*)&Ks[key][dc * 32 + 16 + 8 * hi];
                    s[sub]  = wmma_bf16(aq[dc], kf, s[sub]);
                    kf.q[0] = *(const u32x4*)&Kps[key][dc * 32 + 8 * hi];
                    kf.q[1] = *(const u32x4*)&Kps[key][dc * 32 + 16 + 8 * hi];
                    s[sub]  = wmma_bf16(aqp[dc], kf, s[sub]);
                }
            }

            // online softmax over these 32 columns (scores already scaled)
            float rmax[8], alpha[8], rsum[8];
#pragma unroll
            for (int vv = 0; vv < 8; ++vv)
                rmax[vv] = fmaxf(s[0][vv], s[1][vv]);
#pragma unroll
            for (int off = 1; off < 16; off <<= 1)
#pragma unroll
                for (int vv = 0; vv < 8; ++vv)
                    rmax[vv] = fmaxf(rmax[vv], __shfl_xor(rmax[vv], off, 32));
#pragma unroll
            for (int vv = 0; vv < 8; ++vv) {
                const float nm = fmaxf(m_run[vv], rmax[vv]);
                alpha[vv] = __expf(m_run[vv] - nm);
                m_run[vv] = nm;
                const float p0 = __expf(s[0][vv] - nm);
                const float p1 = __expf(s[1][vv] - nm);
                s[0][vv] = p0; s[1][vv] = p1;
                rsum[vv] = p0 + p1;
            }
#pragma unroll
            for (int off = 1; off < 16; off <<= 1)
#pragma unroll
                for (int vv = 0; vv < 8; ++vv)
                    rsum[vv] += __shfl_xor(rsum[vv], off, 32);
#pragma unroll
            for (int vv = 0; vv < 8; ++vv)
                l_run[vv] = l_run[vv] * alpha[vv] + rsum[vv];
#pragma unroll
            for (int dj = 0; dj < 4; ++dj)
#pragma unroll
                for (int vv = 0; vv < 8; ++vv) o[dj][vv] *= alpha[vv];

            // re-layout P (C/D frag -> A frag) via per-wave LDS staging
#pragma unroll
            for (int vv = 0; vv < 8; ++vv) {
                const int pr = vv + 8 * hi;
                Ps[wave][pr][m]      = f2bf(s[0][vv]);
                Ps[wave][pr][16 + m] = f2bf(s[1][vv]);
            }
            Frag pf;
            pf.q[0] = *(const u32x4*)&Ps[wave][m][8 * hi];
            pf.q[1] = *(const u32x4*)&Ps[wave][m][16 + 8 * hi];

            // O += P @ V  (K-dim = 32 keys)
#pragma unroll
            for (int dj = 0; dj < 4; ++dj) {
                Frag vf;
                const int dd = dj * 16 + m;
                vf.q[0] = *(const u32x4*)&Vt[dd][kb + 8 * hi];
                vf.q[1] = *(const u32x4*)&Vt[dd][kb + 16 + 8 * hi];
                o[dj] = wmma_bf16(pf, vf, o[dj]);
            }
        }
        __syncthreads();
    }

    // normalize and store attention output [b, n, h*64+d] bf16
    float inv[8];
#pragma unroll
    for (int vv = 0; vv < 8; ++vv) inv[vv] = 1.f / l_run[vv];
    const int qrow0 = qt * 128 + wave * 16;
#pragma unroll
    for (int dj = 0; dj < 4; ++dj) {
        const int col = h * DIM_HEAD + dj * 16 + m;
#pragma unroll
        for (int vv = 0; vv < 8; ++vv) {
            const int n = qrow0 + vv + 8 * hi;
            aow[(((size_t)b << 10) + n) * INNER + col] = f2bf(o[dj][vv] * inv[vv]);
        }
    }
}

// ---------------------------------------------------------------------------
// Kernel 3: out = attn_out @ w_out + b_out  (f32). grid = (32 rows, 6 cols).
// A tile (already bf16) staged via async global->LDS DMA (ASYNCcnt).
// ---------------------------------------------------------------------------
__global__ void __launch_bounds__(256)
out_kernel(const u16* __restrict__ ao, const float* __restrict__ w_out,
           const float* __restrict__ b_out, float* __restrict__ out)
{
    __shared__ u16 As[128][40];
    __shared__ u16 Bs[128][40];

    const int r0 = blockIdx.x * 128;
    const int c0 = blockIdx.y * 128;
    const int tid  = threadIdx.x;
    const int lane = tid & 31;
    const int wave = tid >> 5;
    const int wrow = wave >> 1;
    const int wcol = wave & 1;
    const int m    = lane & 15;
    const int hi   = lane >> 4;

    v8f acc[2][4];
    const v8f zero = {0.f,0.f,0.f,0.f,0.f,0.f,0.f,0.f};
#pragma unroll
    for (int i = 0; i < 2; ++i)
#pragma unroll
        for (int j = 0; j < 4; ++j) acc[i][j] = zero;

    for (int k0 = 0; k0 < INNER; k0 += 32) {
        {
            const int row = tid >> 1;
            const int kk  = (tid & 1) * 16;
            const size_t sb = (size_t)(r0 + row) * INNER + k0 + kk;
            async_copy_b128((unsigned)(size_t)&As[row][kk],     &ao[sb]);
            async_copy_b128((unsigned)(size_t)&As[row][kk + 8], &ao[sb + 8]);
        }
        {
            const int kk = tid >> 3;
            const int n  = (tid & 7) * 16;
            const float* src = &w_out[(size_t)(k0 + kk) * DIM + c0 + n];
#pragma unroll
            for (int j = 0; j < 16; ++j) Bs[n + j][kk] = f2bf(src[j]);
        }
        wait_async0();
        __syncthreads();

        Frag af[2];
#pragma unroll
        for (int i = 0; i < 2; ++i) {
            const int ar = wrow * 32 + i * 16 + m;
            af[i].q[0] = *(const u32x4*)&As[ar][8 * hi];
            af[i].q[1] = *(const u32x4*)&As[ar][16 + 8 * hi];
        }
#pragma unroll
        for (int j = 0; j < 4; ++j) {
            Frag bf;
            const int bn = wcol * 64 + j * 16 + m;
            bf.q[0] = *(const u32x4*)&Bs[bn][8 * hi];
            bf.q[1] = *(const u32x4*)&Bs[bn][16 + 8 * hi];
#pragma unroll
            for (int i = 0; i < 2; ++i) acc[i][j] = wmma_bf16(af[i], bf, acc[i][j]);
        }
        __syncthreads();
    }

#pragma unroll
    for (int j = 0; j < 4; ++j) {
        const int c = c0 + wcol * 64 + j * 16 + m;
        const float bias = b_out[c];
#pragma unroll
        for (int i = 0; i < 2; ++i) {
#pragma unroll
            for (int vv = 0; vv < 8; ++vv) {
                const int r = r0 + wrow * 32 + i * 16 + vv + 8 * hi;
                out[(size_t)r * DIM + c] = acc[i][j][vv] + bias;
            }
        }
    }
}

// ---------------------------------------------------------------------------
extern "C" void kernel_launch(void* const* d_in, const int* in_sizes, int n_in,
                              void* d_out, int out_size, void* d_ws, size_t ws_size,
                              hipStream_t stream) {
    const float* x       = (const float*)d_in[0];
    const float* pos     = (const float*)d_in[1];
    const float* w_qkv   = (const float*)d_in[2];
    const float* w_qkpos = (const float*)d_in[3];
    const float* w_out   = (const float*)d_in[4];
    const float* b_out   = (const float*)d_in[5];
    float* out = (float*)d_out;

    const size_t TSZ = (size_t)BATCH * HEADS * SEQ * DIM_HEAD;   // 3145728
    u16* qw  = (u16*)d_ws;
    u16* kw  = qw  + TSZ;
    u16* vw  = kw  + TSZ;
    u16* qpw = vw  + TSZ;
    u16* kpw = qpw + TSZ;
    u16* aow = kpw + TSZ;   // BATCH*SEQ*INNER == TSZ elements

    proj_kernel<<<dim3(32, 30), 256, 0, stream>>>(x, pos, w_qkv, w_qkpos,
                                                  qw, kw, vw, qpw, kpw);
    attn_kernel<<<dim3(8, BATCH * HEADS), 256, 0, stream>>>(qw, kw, vw, qpw, kpw, aow);
    out_kernel<<<dim3(32, 6), 256, 0, stream>>>(aow, w_out, b_out, out);
}
